// SelectiveScan_8572754723195
// MI455X (gfx1250) — matrix operand order repaired and checked by reading, not compiled, not run on hardware
//
#include <hip/hip_runtime.h>
#include <hip/hip_bf16.h>
#include <math.h>

// ---------------------------------------------------------------------------
// Selective scan (Mamba block) for MI455X / gfx1250, wave32 + WMMA bf16.
// B=2, L=2048, D=1024, N=16, R=64(dt_rank), P=96(R+2N), K_conv=4, T=B*L=4096
//
// Roofline: 3 big GEMMs (4096x1024x1024) = ~26 GFLOP, ~150MB HBM traffic
// (~7us @ 23.3 TB/s) -> GEMMs via bf16 WMMA, memory-bound. The recurrence is
// the latency bottleneck, so it is chunk-parallelized (assoc. combine
// (a1,b1)o(a2,b2)=(a1a2, a2*b1+b2)): 32 chunks x 64 steps, 3 phases.
// ---------------------------------------------------------------------------

#define TT 4096   // total tokens B*L
#define DD 1024   // embed dim
#define NN 16     // state dim
#define RR 64     // dt rank
#define PP 96     // R + 2N
#define LL 2048   // sequence length
#define BB 2      // batch
#define CCH 32    // scan chunks per sequence
#define LC 64     // tokens per chunk (LL/CCH)

typedef __attribute__((ext_vector_type(16))) __bf16 v16bf;
typedef __attribute__((ext_vector_type(8)))  __bf16 v8bf;
typedef __attribute__((ext_vector_type(8)))  float  v8f;

__device__ __forceinline__ __bf16 to_bf16(float f) {
  union { float f; unsigned u; } v; v.f = f;
  unsigned r = v.u + 0x7FFFu + ((v.u >> 16) & 1u);   // round-to-nearest-even
  unsigned short h = (unsigned short)(r >> 16);
  union { unsigned short s; __bf16 b; } o; o.s = h;
  return o.b;
}

// ---- CDNA5 async direct-to-LDS copy (ASYNCcnt path), per ISA ch.10/15.18 ---
__device__ __forceinline__ void async_load_b128_to_lds(unsigned lds_off,
                                                       const void* gaddr) {
  asm volatile("global_load_async_to_lds_b128 %0, %1, off"
               :: "v"(lds_off), "v"((unsigned long long)(size_t)gaddr)
               : "memory");
}
__device__ __forceinline__ void wait_asynccnt0() {
  asm volatile("s_wait_asynccnt 0x0" ::: "memory");
}

// --------------------------- f32 -> bf16 copy ------------------------------
__global__ void cvt_bf16_kernel(const float* __restrict__ in,
                                __bf16* __restrict__ out, int n) {
  int i = blockIdx.x * blockDim.x + threadIdx.x;
  if (i < n) out[i] = to_bf16(in[i]);
}

// ---------------- fragment gather helpers (ISA 7.12.2 layouts) -------------
__device__ __forceinline__ v16bf load_a_frag(const __bf16* __restrict__ Arow,
                                             int k0, int g) {
  v8bf a0 = *(const v8bf*)(Arow + k0 + g * 8);        // K = k0+g*8+e
  v8bf a1 = *(const v8bf*)(Arow + k0 + 16 + g * 8);   // K = k0+16+g*8+e
  return __builtin_shufflevector(a0, a1,
      0, 1, 2, 3, 4, 5, 6, 7, 8, 9, 10, 11, 12, 13, 14, 15);
}
__device__ __forceinline__ v16bf load_b_frag(const __bf16* __restrict__ Wrow,
                                             int k0, int g) {
  v8bf b0 = *(const v8bf*)(Wrow + k0 + g * 8);        // K = k0+g*8+e
  v8bf b1 = *(const v8bf*)(Wrow + k0 + 16 + g * 8);   // K = k0+16+g*8+e
  return __builtin_shufflevector(b0, b1,
      0, 1, 2, 3, 4, 5, 6, 7, 8, 9, 10, 11, 12, 13, 14, 15);
}

// -------- WMMA GEMM, 2x2 tiles/wave: C[T,O] = A[T,K] * W[O,K]^T ------------
// 256 threads = 8 waves in 2x4; each wave owns a 32x32 tile (4 WMMA accs,
// A/B fragment reuse -> 2 b128 loads per WMMA). Block tile 64 x 128.
// Requires T%64==0, O%128==0, K%32==0.
__global__ void gemm_bf16_wmma2x2(const __bf16* __restrict__ A,
                                  const __bf16* __restrict__ W,
                                  float* __restrict__ C,
                                  int T, int O, int K) {
  const int lane = threadIdx.x & 31;
  const int wave = threadIdx.x >> 5;
  const int g = lane >> 4;
  const int m = lane & 15;
  const int t0 = blockIdx.y * 64 + (wave >> 2) * 32;
  const int o0 = blockIdx.x * 128 + (wave & 3) * 32;

  v8f acc00 = {}, acc01 = {}, acc10 = {}, acc11 = {};
  const __bf16* Arow0 = A + (size_t)(t0 + m) * K;
  const __bf16* Arow1 = A + (size_t)(t0 + 16 + m) * K;
  const __bf16* Wrow0 = W + (size_t)(o0 + m) * K;
  const __bf16* Wrow1 = W + (size_t)(o0 + 16 + m) * K;

  for (int k0 = 0; k0 < K; k0 += 32) {
    v16bf a0 = load_a_frag(Arow0, k0, g);
    v16bf a1 = load_a_frag(Arow1, k0, g);
    v16bf b0 = load_b_frag(Wrow0, k0, g);
    v16bf b1 = load_b_frag(Wrow1, k0, g);
    acc00 = __builtin_amdgcn_wmma_f32_16x16x32_bf16(false, a0, false, b0, (short)0, acc00, false, false);
    acc01 = __builtin_amdgcn_wmma_f32_16x16x32_bf16(false, a0, false, b1, (short)0, acc01, false, false);
    acc10 = __builtin_amdgcn_wmma_f32_16x16x32_bf16(false, a1, false, b0, (short)0, acc10, false, false);
    acc11 = __builtin_amdgcn_wmma_f32_16x16x32_bf16(false, a1, false, b1, (short)0, acc11, false, false);
  }

  // D element r -> row M = g*8+r, col N = m (per 16x16 f32 C/D layout)
  float* C00 = C + (size_t)(t0 + g * 8) * O + (o0 + m);
  float* C10 = C + (size_t)(t0 + 16 + g * 8) * O + (o0 + m);
#pragma unroll
  for (int r = 0; r < 8; ++r) {
    C00[(size_t)r * O]      = acc00[r];
    C00[(size_t)r * O + 16] = acc01[r];
    C10[(size_t)r * O]      = acc10[r];
    C10[(size_t)r * O + 16] = acc11[r];
  }
}

// ----------- simple WMMA GEMM (for O=96): block tile 64x32 -----------------
__global__ void gemm_bf16_wmma(const __bf16* __restrict__ A,
                               const __bf16* __restrict__ W,
                               float* __restrict__ C,
                               int T, int O, int K) {
  const int lane = threadIdx.x & 31;
  const int wave = threadIdx.x >> 5;
  const int g = lane >> 4;
  const int m = lane & 15;
  const int t0 = blockIdx.y * 64 + (wave >> 1) * 16;
  const int o0 = blockIdx.x * 32 + (wave & 1) * 16;

  v8f acc = {};
  const __bf16* Arow = A + (size_t)(t0 + m) * K;
  const __bf16* Wrow = W + (size_t)(o0 + m) * K;
  for (int k0 = 0; k0 < K; k0 += 32) {
    v16bf a = load_a_frag(Arow, k0, g);
    v16bf b = load_b_frag(Wrow, k0, g);
    acc = __builtin_amdgcn_wmma_f32_16x16x32_bf16(false, a, false, b, (short)0, acc, false, false);
  }
  float* Cp = C + (size_t)(t0 + g * 8) * O + (o0 + m);
#pragma unroll
  for (int r = 0; r < 8; ++r) Cp[(size_t)r * O] = acc[r];
}

// ----------------- extract dt_un (params[:, :R]) as bf16 -------------------
__global__ void extract_dtun_kernel(const float* __restrict__ params,
                                    __bf16* __restrict__ dtun) {
  int i = blockIdx.x * blockDim.x + threadIdx.x;
  if (i < TT * RR) {
    int t = i / RR, r = i % RR;
    dtun[i] = to_bf16(params[(size_t)t * PP + r]);
  }
}

// ------------- depthwise causal conv (K=4) along L + SiLU -> u -------------
__global__ void conv_silu_kernel(const float* __restrict__ xp,   // [T,D]
                                 const float* __restrict__ cw,   // [D,1,4]
                                 float* __restrict__ u) {        // [T,D]
  int i = blockIdx.x * blockDim.x + threadIdx.x;
  if (i >= TT * DD) return;
  int d = i & (DD - 1);
  int t = i >> 10;          // DD = 1024
  int l = t & (LL - 1);
  float acc = 0.f;
#pragma unroll
  for (int k = 0; k < 4; ++k) {
    int ll = l - 3 + k;
    if (ll >= 0) acc += cw[d * 4 + k] * xp[(size_t)(t - 3 + k) * DD + d];
  }
  u[i] = acc / (1.f + expf(-acc));   // SiLU
}

// ------- stage one chunk's B_proj/C_proj (LC tokens x 32 f32) to LDS -------
// Uses CDNA5 async-to-LDS (b128); 512 segments of 16B, 2 per thread.
__device__ __forceinline__ void stage_chunk_params(
    float (&sP)[LC][32], const float* __restrict__ params, int t0c, int tid) {
#pragma unroll
  for (int rep = 0; rep < 2; ++rep) {
    int idx = rep * 256 + tid;        // 0..511
    int tok = idx >> 3;               // 0..63
    int seg = idx & 7;                // 0..7 (4 floats each)
    const void* g = params + (size_t)(t0c + tok) * PP + RR + seg * 4;
    unsigned lds_off = (unsigned)(size_t)&sP[tok][seg * 4];
    async_load_b128_to_lds(lds_off, g);
  }
  wait_asynccnt0();
  __syncthreads();
}

// ----------------------- chunked selective scan ----------------------------
// Phase 1: per (b,c,d) compute chunk-local (prod a, local scan end).
// grid = BB*CCH*(DD/256) blocks of 256 threads.
__global__ void scan_phase1(const float* __restrict__ dtlin,  // [T,D]
                            const float* __restrict__ bdt,    // [D]
                            const float* __restrict__ A_log,  // [D,N]
                            const float* __restrict__ params, // [T,P]
                            const float* __restrict__ u,      // [T,D]
                            float* __restrict__ Ap,           // [B,C,N,D]
                            float* __restrict__ He) {         // [B,C,N,D]
  const int dblk = blockIdx.x & 3;
  const int c = (blockIdx.x >> 2) & (CCH - 1);
  const int b = blockIdx.x >> 7;
  const int d = (dblk << 8) + threadIdx.x;
  const int t0c = b * LL + c * LC;

  __shared__ float sP[LC][32];
  stage_chunk_params(sP, params, t0c, threadIdx.x);

  float Acoef[NN], aprod[NN], hend[NN];
#pragma unroll
  for (int n = 0; n < NN; ++n) {
    Acoef[n] = -expf(A_log[d * NN + n]);
    aprod[n] = 1.f; hend[n] = 0.f;
  }
  const float bd = bdt[d];

  for (int l = 0; l < LC; ++l) {
    const size_t t = (size_t)t0c + l;
    float dtv = dtlin[t * DD + d] + bd;
    dtv = (dtv > 20.f) ? dtv : log1pf(expf(dtv));   // softplus
    const float uu = u[t * DD + d];
#pragma unroll
    for (int n = 0; n < NN; ++n) {
      const float a = expf(Acoef[n] * dtv);
      const float btil = (a - 1.f) / (Acoef[n] + 1e-8f);
      aprod[n] *= a;
      hend[n] = a * hend[n] + btil * sP[l][n] * uu;
    }
  }
  const size_t base = ((size_t)(b * CCH + c) * NN) * DD + d;
#pragma unroll
  for (int n = 0; n < NN; ++n) {
    Ap[base + (size_t)n * DD] = aprod[n];
    He[base + (size_t)n * DD] = hend[n];
  }
}

// Phase 2: per (b,d) sequential combine across 32 chunks -> chunk init states
__global__ void scan_phase2(const float* __restrict__ Ap,
                            const float* __restrict__ He,
                            float* __restrict__ Hinit) {      // [B,C,N,D]
  const int i = blockIdx.x * blockDim.x + threadIdx.x;  // 0..2047
  const int b = i >> 10;
  const int d = i & (DD - 1);
  float h[NN];
#pragma unroll
  for (int n = 0; n < NN; ++n) h[n] = 0.f;
  for (int c = 0; c < CCH; ++c) {
    const size_t base = ((size_t)(b * CCH + c) * NN) * DD + d;
#pragma unroll
    for (int n = 0; n < NN; ++n) {
      const size_t o = base + (size_t)n * DD;
      Hinit[o] = h[n];
      h[n] = Ap[o] * h[n] + He[o];
    }
  }
}

// Phase 3: replay each chunk with correct init state, produce gated y (bf16)
__global__ void scan_phase3(const float* __restrict__ dtlin,
                            const float* __restrict__ bdt,
                            const float* __restrict__ A_log,
                            const float* __restrict__ params,
                            const float* __restrict__ u,
                            const float* __restrict__ z,
                            const float* __restrict__ Dp,
                            const float* __restrict__ Hinit,
                            __bf16* __restrict__ ybf) {       // [T,D]
  const int dblk = blockIdx.x & 3;
  const int c = (blockIdx.x >> 2) & (CCH - 1);
  const int b = blockIdx.x >> 7;
  const int d = (dblk << 8) + threadIdx.x;
  const int t0c = b * LL + c * LC;

  __shared__ float sP[LC][32];
  stage_chunk_params(sP, params, t0c, threadIdx.x);

  float Acoef[NN], h[NN];
  const size_t base = ((size_t)(b * CCH + c) * NN) * DD + d;
#pragma unroll
  for (int n = 0; n < NN; ++n) {
    Acoef[n] = -expf(A_log[d * NN + n]);
    h[n] = Hinit[base + (size_t)n * DD];
  }
  const float bd = bdt[d];
  const float dp = Dp[d];

  for (int l = 0; l < LC; ++l) {
    const size_t t = (size_t)t0c + l;
    float dtv = dtlin[t * DD + d] + bd;
    dtv = (dtv > 20.f) ? dtv : log1pf(expf(dtv));
    const float uu = u[t * DD + d];
    const float zz = z[t * DD + d];
    float y = 0.f;
#pragma unroll
    for (int n = 0; n < NN; ++n) {
      const float a = expf(Acoef[n] * dtv);
      const float btil = (a - 1.f) / (Acoef[n] + 1e-8f);
      h[n] = a * h[n] + btil * sP[l][n] * uu;
      y += sP[l][16 + n] * h[n];
    }
    y = (y + uu * dp) * (zz / (1.f + expf(-zz)));
    ybf[t * DD + d] = to_bf16(y);
  }
}

// ---------------------------------------------------------------------------
extern "C" void kernel_launch(void* const* d_in, const int* in_sizes, int n_in,
                              void* d_out, int out_size, void* d_ws, size_t ws_size,
                              hipStream_t stream) {
  (void)in_sizes; (void)n_in; (void)out_size; (void)ws_size;

  const float* x     = (const float*)d_in[0];  // [2,2048,1024]
  const float* Wx    = (const float*)d_in[1];  // [1024,1024]
  const float* Wz    = (const float*)d_in[2];  // [1024,1024]
  const float* Wp    = (const float*)d_in[3];  // [96,1024]
  const float* convw = (const float*)d_in[4];  // [1024,1,4]
  const float* Wdt   = (const float*)d_in[5];  // [1024,64]
  const float* bdt   = (const float*)d_in[6];  // [1024]
  const float* A_log = (const float*)d_in[7];  // [1024,16]
  const float* Dp    = (const float*)d_in[8];  // [1024]
  const float* Wo    = (const float*)d_in[9];  // [1024,1024]
  float* out = (float*)d_out;                  // [2,2048,1024] f32

  char* ws = (char*)d_ws;
  size_t off = 0;
  auto alloc = [&](size_t bytes) -> void* {
    void* p = ws + off;
    off = (off + bytes + 255) & ~(size_t)255;
    return p;
  };

  __bf16* xb    = (__bf16*)alloc((size_t)TT * DD * 2);
  __bf16* Wxb   = (__bf16*)alloc((size_t)DD * DD * 2);
  __bf16* Wzb   = (__bf16*)alloc((size_t)DD * DD * 2);
  __bf16* Wpb   = (__bf16*)alloc((size_t)PP * DD * 2);
  __bf16* Wdtb  = (__bf16*)alloc((size_t)DD * RR * 2);
  __bf16* Wob   = (__bf16*)alloc((size_t)DD * DD * 2);
  __bf16* dtunb = (__bf16*)alloc((size_t)TT * RR * 2);
  __bf16* ybf   = (__bf16*)alloc((size_t)TT * DD * 2);
  float* xproj  = (float*)alloc((size_t)TT * DD * 4);
  float* zbuf   = (float*)alloc((size_t)TT * DD * 4);
  float* paramsb= (float*)alloc((size_t)TT * PP * 4);
  float* dtlin  = (float*)alloc((size_t)TT * DD * 4);
  float* ubuf   = (float*)alloc((size_t)TT * DD * 4);
  float* Ap     = (float*)alloc((size_t)BB * CCH * NN * DD * 4);
  float* He     = (float*)alloc((size_t)BB * CCH * NN * DD * 4);
  float* Hinit  = (float*)alloc((size_t)BB * CCH * NN * DD * 4);

  const int CT = 256;

  // 1. bf16 conversions
  cvt_bf16_kernel<<<(TT * DD + CT - 1) / CT, CT, 0, stream>>>(x, xb, TT * DD);
  cvt_bf16_kernel<<<(DD * DD + CT - 1) / CT, CT, 0, stream>>>(Wx, Wxb, DD * DD);
  cvt_bf16_kernel<<<(DD * DD + CT - 1) / CT, CT, 0, stream>>>(Wz, Wzb, DD * DD);
  cvt_bf16_kernel<<<(PP * DD + CT - 1) / CT, CT, 0, stream>>>(Wp, Wpb, PP * DD);
  cvt_bf16_kernel<<<(DD * RR + CT - 1) / CT, CT, 0, stream>>>(Wdt, Wdtb, DD * RR);
  cvt_bf16_kernel<<<(DD * DD + CT - 1) / CT, CT, 0, stream>>>(Wo, Wob, DD * DD);

  // 2. projections: x_proj = x Wx^T, z = x Wz^T, params = x Wp^T
  dim3 gBig(DD / 128, TT / 64);
  dim3 gP(PP / 32, TT / 64);
  gemm_bf16_wmma2x2<<<gBig, 256, 0, stream>>>(xb, Wxb, xproj, TT, DD, DD);
  gemm_bf16_wmma2x2<<<gBig, 256, 0, stream>>>(xb, Wzb, zbuf, TT, DD, DD);
  gemm_bf16_wmma<<<gP, 256, 0, stream>>>(xb, Wpb, paramsb, TT, PP, DD);

  // 3. dt linear: dtlin = dt_un Wdt^T
  extract_dtun_kernel<<<(TT * RR + CT - 1) / CT, CT, 0, stream>>>(paramsb, dtunb);
  gemm_bf16_wmma2x2<<<gBig, 256, 0, stream>>>(dtunb, Wdtb, dtlin, TT, DD, RR);

  // 4. depthwise causal conv + SiLU -> u
  conv_silu_kernel<<<(TT * DD + CT - 1) / CT, CT, 0, stream>>>(xproj, convw, ubuf);

  // 5. chunk-parallel selective scan
  scan_phase1<<<BB * CCH * (DD / 256), 256, 0, stream>>>(dtlin, bdt, A_log,
                                                         paramsb, ubuf, Ap, He);
  scan_phase2<<<(BB * DD) / 256, 256, 0, stream>>>(Ap, He, Hinit);
  scan_phase3<<<BB * CCH * (DD / 256), 256, 0, stream>>>(dtlin, bdt, A_log,
                                                         paramsb, ubuf, zbuf,
                                                         Dp, Hinit, ybf);

  // 6. output projection: out = y Wo^T (f32 accumulate/store)
  gemm_bf16_wmma2x2<<<gBig, 256, 0, stream>>>(ybf, Wob, out, TT, DD, DD);
}
